// NodeClassificationMPNSelfAttention_79568564126377
// MI455X (gfx1250) — matrix operand order repaired and checked by reading, not compiled, not run on hardware
//
#include <hip/hip_runtime.h>

typedef _Float16 half_t;
typedef __attribute__((ext_vector_type(16))) _Float16 v16h;
typedef __attribute__((ext_vector_type(8)))  _Float16 v8h;
typedef __attribute__((ext_vector_type(8)))  float    v8f;

#define NN   2048
#define EE   65536
#define BB   4
#define PP   4096
#define DD   128
#define EDD  128
#define HIDD 256

union FragU { v16h v; v8h h[2]; };

__device__ __forceinline__ v8h ld8(const half_t* p) { return *(const v8h*)p; }

// ---------------------------------------------------------------------------
// Specialized gathered-A WMMA GEMM:  C[M,N] = relu(A_virtual[M,K] @ W[N,K]^T + b)
// A virtual row = [seg0 (gather idx0) | seg1 (gather idx1 / identity) | seg2 | 0]
// Segment boundaries S1,S2 and K are template constants, all multiples of 16,
// so with the K-loop fully unrolled every segment select folds at compile time
// and the inner loop is pure b128 loads (immediate offsets) + 8 WMMAs.
// One wave per block, 32(M) x 64(N) tile, grid = (M/32, N/64).
// OUTMODE: 0 = store f32; 1 = store f32 + f16; 2 = store f16;
//          3 = atomicAdd f32 into outF32[aidx[row]*N + col]   (segment_sum)
// ---------------------------------------------------------------------------
template<int K, int S1, int S2, bool G0, bool G1, bool HAS1, bool HAS2, int OUTMODE>
__global__ __launch_bounds__(32) void k_gemm(
    const half_t* __restrict__ p0, int st0, const int* __restrict__ idx0,
    const half_t* __restrict__ p1, int st1, const int* __restrict__ idx1,
    const half_t* __restrict__ p2, int st2,
    const half_t* __restrict__ W, const float* __restrict__ bias, int N,
    float* __restrict__ outF32, half_t* __restrict__ outF16,
    const int* __restrict__ aidx)
{
  int lane = threadIdx.x;
  int hl = lane >> 4;          // lane half selects K sub-chunk per ISA layout
  int r  = lane & 15;
  int mbase = blockIdx.x * 32;
  int nbase = blockIdx.y * 64;

  // per-lane gathered row base pointers (rows m = r and 16+r), hl*8 pre-added
  const half_t* A0[2]; const half_t* A1[2]; const half_t* A2[2];
#pragma unroll
  for (int mi = 0; mi < 2; ++mi) {
    int e = mbase + mi * 16 + r;
    A0[mi] = p0 + (size_t)(G0 ? idx0[e] : e) * st0 + hl * 8;
    A1[mi] = HAS1 ? p1 + (size_t)(G1 ? idx1[e] : e) * st1 + hl * 8 : (const half_t*)nullptr;
    A2[mi] = HAS2 ? p2 + (size_t)e * st2 + hl * 8 : (const half_t*)nullptr;
  }
  // B operand: lane holds W[n = nbase+ni*16+r, kb + hl*8 + {0..7, 16..23}]
  const half_t* Wr[4]; float bv[4];
#pragma unroll
  for (int ni = 0; ni < 4; ++ni) {
    int n = nbase + ni * 16 + r;
    Wr[ni] = W + (size_t)n * K + hl * 8;
    bv[ni] = bias[n];
  }

  v8f acc[2][4];
  v8f zf = {};
#pragma unroll
  for (int mi = 0; mi < 2; ++mi)
#pragma unroll
    for (int ni = 0; ni < 4; ++ni) acc[mi][ni] = zf;

#pragma unroll
  for (int kb = 0; kb < K; kb += 32) {
    FragU a[2], bf[4];
#pragma unroll
    for (int mi = 0; mi < 2; ++mi) {
#pragma unroll
      for (int c = 0; c < 2; ++c) {
        const int kc = kb + c * 16;    // compile-time: 16-aligned chunk window
        v8h val;
        if (kc < S1)                 val = ld8(A0[mi] + kc);
        else if (HAS1 && kc < S2)    val = ld8(A1[mi] + (kc - S1));
        else if (HAS2)               val = ld8(A2[mi] + (kc - S2));
        else                         { v8h z = {}; val = z; }
        a[mi].h[c] = val;
      }
    }
#pragma unroll
    for (int ni = 0; ni < 4; ++ni) {
      bf[ni].h[0] = ld8(Wr[ni] + kb);
      bf[ni].h[1] = ld8(Wr[ni] + kb + 16);
    }
#pragma unroll
    for (int mi = 0; mi < 2; ++mi)
#pragma unroll
      for (int ni = 0; ni < 4; ++ni)
        acc[mi][ni] = __builtin_amdgcn_wmma_f32_16x16x32_f16(
            false, a[mi].v, false, bf[ni].v, (short)0, acc[mi][ni], false, false);
  }

  // epilogue: C layout VGPR rr -> row rr + 8*hl, col = r
#pragma unroll
  for (int mi = 0; mi < 2; ++mi) {
#pragma unroll
    for (int rr = 0; rr < 8; ++rr) {
      int row = mbase + mi * 16 + rr + 8 * hl;
      int arow = (OUTMODE == 3) ? aidx[row] : row;
#pragma unroll
      for (int ni = 0; ni < 4; ++ni) {
        int col = nbase + ni * 16 + r;
        float val = fmaxf(acc[mi][ni][rr] + bv[ni], 0.0f);
        if (OUTMODE == 3) {
          atomicAdd(&outF32[(size_t)arow * N + col], val);
        } else {
          if (OUTMODE == 0 || OUTMODE == 1) outF32[(size_t)row * N + col] = val;
          if (OUTMODE == 1 || OUTMODE == 2) outF16[(size_t)row * N + col] = (half_t)val;
        }
      }
    }
  }
}

// ---------------------------------------------------------------------------
// Small helper kernels
// ---------------------------------------------------------------------------
__global__ void k_f32_to_f16(const float* __restrict__ in, half_t* __restrict__ out, int n) {
  int i = blockIdx.x * blockDim.x + threadIdx.x;
  if (i < n) out[i] = (half_t)in[i];
}

__global__ void k_f32_to_f16_pad(const float* __restrict__ in, half_t* __restrict__ out,
                                 int rows, int kin, int kout) {
  int i = blockIdx.x * blockDim.x + threadIdx.x;
  if (i >= rows * kout) return;
  int rr = i / kout, k = i - rr * kout;
  out[i] = (k < kin) ? (half_t)in[rr * kin + k] : (half_t)0.0f;
}

// pack node_features f32 [N,128] into nf16 [N,144] cols 0..127 (res occupies 128..143)
__global__ void k_pack_nf(const float* __restrict__ nf, half_t* __restrict__ nf16) {
  int i = blockIdx.x * blockDim.x + threadIdx.x;   // NN*128
  int n = i >> 7, c = i & 127;
  nf16[(size_t)n * 144 + c] = (half_t)nf[i];
}

// queries/values projection: q[b,o,p] = q_b[o] + sum_c q_w[o,c]*fm[b,c,p]
__global__ void k_qv(const float* __restrict__ fm,
                     const float* __restrict__ qw, const float* __restrict__ qb,
                     const float* __restrict__ vw, const float* __restrict__ vb,
                     float* __restrict__ q, float* __restrict__ v) {
  int i = blockIdx.x * blockDim.x + threadIdx.x;   // BB*16*PP
  int p = i & (PP - 1);
  int o = (i >> 12) & 15;
  int b = i >> 16;
  const float* f = fm + ((size_t)b * 32) * PP + p;
  float qa = qb[o], va = vb[o];
#pragma unroll
  for (int c = 0; c < 32; ++c) {
    float x = f[(size_t)c * PP];
    qa += qw[o * 32 + c] * x;
    va += vw[o * 32 + c] * x;
  }
  q[i] = qa; v[i] = va;
}

// per-node attention: only the node's own batch matters (result[batch_index[n],:,n]).
// One 256-thread block per node; writes res (f16) into nf16[:,128:144].
__global__ __launch_bounds__(256) void k_attention(
    const float* __restrict__ nf, const float* __restrict__ key_w, const float* __restrict__ key_b,
    const float* __restrict__ q, const float* __restrict__ v,
    const int* __restrict__ bix, half_t* __restrict__ nf16)
{
  int n = blockIdx.x;
  int t = threadIdx.x;
  __shared__ float kvec[16];
  __shared__ float red[256];
  __shared__ float racc[16];
  int b = bix[n];
  if (t < 16) {
    float a = key_b[t];
    const float* row = nf + (size_t)n * DD;
    const float* kw = key_w + t * DD;
    for (int c = 0; c < DD; ++c) a += row[c] * kw[c];
    kvec[t] = a;
    racc[t] = 0.0f;
  }
  __syncthreads();
  float kk[16];
#pragma unroll
  for (int j = 0; j < 16; ++j) kk[j] = kvec[j];
  const float* qb_ = q + (size_t)b * 16 * PP;
  const float* vb_ = v + (size_t)b * 16 * PP;
  float logit[16];
  float lmax = -3.4e38f;
#pragma unroll
  for (int i = 0; i < 16; ++i) {
    int p = t + i * 256;
    float a = 0.0f;
#pragma unroll
    for (int j = 0; j < 16; ++j) a += kk[j] * qb_[(size_t)j * PP + p];
    logit[i] = a;
    lmax = fmaxf(lmax, a);
  }
  red[t] = lmax; __syncthreads();
  for (int s = 128; s > 0; s >>= 1) { if (t < s) red[t] = fmaxf(red[t], red[t + s]); __syncthreads(); }
  float gmax = red[0];
  __syncthreads();
  float w[16]; float lsum = 0.0f;
#pragma unroll
  for (int i = 0; i < 16; ++i) { w[i] = expf(logit[i] - gmax); lsum += w[i]; }
  red[t] = lsum; __syncthreads();
  for (int s = 128; s > 0; s >>= 1) { if (t < s) red[t] += red[t + s]; __syncthreads(); }
  float gsum = red[0];
  float part[16];
#pragma unroll
  for (int j = 0; j < 16; ++j) part[j] = 0.0f;
#pragma unroll
  for (int i = 0; i < 16; ++i) {
    int p = t + i * 256;
    float wp = w[i];
#pragma unroll
    for (int j = 0; j < 16; ++j) part[j] += wp * vb_[(size_t)j * PP + p];
  }
#pragma unroll
  for (int j = 0; j < 16; ++j) atomicAdd(&racc[j], part[j]);
  __syncthreads();
  if (t < 16) nf16[(size_t)n * 144 + 128 + t] = (half_t)(racc[t] / gsum);
}

// edge readout: pe[e] = ec_b2 + sum_j relu(ec_b1[j] + ef[e,:].ec_w1[j,:]) * ec_w2[j]
__global__ __launch_bounds__(128) void k_edge_readout(
    const float* __restrict__ ef, const float* __restrict__ w1, const float* __restrict__ b1,
    const float* __restrict__ w2, const float* __restrict__ b2, float* __restrict__ pe)
{
  int e = blockIdx.x * blockDim.x + threadIdx.x;
  if (e >= EE) return;
  float rv[EDD];
  const float* row = ef + (size_t)e * EDD;
#pragma unroll
  for (int c = 0; c < EDD; ++c) rv[c] = row[c];
  float out = b2[0];
  for (int j = 0; j < 64; ++j) {
    const float* wr = w1 + j * EDD;
    float a = b1[j];
#pragma unroll
    for (int c = 0; c < EDD; ++c) a += rv[c] * wr[c];
    out += fmaxf(a, 0.0f) * w2[j];
  }
  pe[e] = out;
}

// node readouts pn (scalar) and pc (17 classes)
__global__ __launch_bounds__(128) void k_node_readout(
    const float* __restrict__ nf,
    const float* __restrict__ nw1, const float* __restrict__ nb1,
    const float* __restrict__ nw2, const float* __restrict__ nb2,
    const float* __restrict__ cw1, const float* __restrict__ cb1,
    const float* __restrict__ cw2, const float* __restrict__ cb2,
    float* __restrict__ pn, float* __restrict__ pc)
{
  int n = blockIdx.x * blockDim.x + threadIdx.x;
  if (n >= NN) return;
  float rv[DD];
  const float* row = nf + (size_t)n * DD;
#pragma unroll
  for (int c = 0; c < DD; ++c) rv[c] = row[c];
  float pna = nb2[0];
  for (int j = 0; j < 64; ++j) {
    const float* wr = nw1 + j * DD;
    float a = nb1[j];
#pragma unroll
    for (int c = 0; c < DD; ++c) a += rv[c] * wr[c];
    pna += fmaxf(a, 0.0f) * nw2[j];
  }
  pn[n] = pna;
  float pca[17];
#pragma unroll
  for (int o = 0; o < 17; ++o) pca[o] = cb2[o];
  for (int j = 0; j < 64; ++j) {
    const float* wr = cw1 + j * DD;
    float a = cb1[j];
#pragma unroll
    for (int c = 0; c < DD; ++c) a += rv[c] * wr[c];
    a = fmaxf(a, 0.0f);
#pragma unroll
    for (int o = 0; o < 17; ++o) pca[o] += a * cw2[o * 64 + j];
  }
#pragma unroll
  for (int o = 0; o < 17; ++o) pc[(size_t)n * 17 + o] = pca[o];
}

// ---------------------------------------------------------------------------
extern "C" void kernel_launch(void* const* d_in, const int* in_sizes, int n_in,
                              void* d_out, int out_size, void* d_ws, size_t ws_size,
                              hipStream_t stream)
{
  (void)in_sizes; (void)n_in; (void)out_size; (void)ws_size;
  const float* x    = (const float*)d_in[0];
  const float* ea   = (const float*)d_in[1];
  const int*   ei   = (const int*)d_in[2];
  const float* fm   = (const float*)d_in[3];
  const int*   bix  = (const int*)d_in[4];
  const float* ne_w = (const float*)d_in[5];  const float* ne_b = (const float*)d_in[6];
  const float* ee_w = (const float*)d_in[7];  const float* ee_b = (const float*)d_in[8];
  const float* me_w1= (const float*)d_in[9];  const float* me_b1= (const float*)d_in[10];
  const float* me_w2= (const float*)d_in[11]; const float* me_b2= (const float*)d_in[12];
  const float* mn_w = (const float*)d_in[13]; const float* mn_b = (const float*)d_in[14];
  const float* key_w= (const float*)d_in[15]; const float* key_b= (const float*)d_in[16];
  const float* q_w  = (const float*)d_in[17]; const float* q_b  = (const float*)d_in[18];
  const float* v_w  = (const float*)d_in[19]; const float* v_b  = (const float*)d_in[20];
  const float* ec_w1= (const float*)d_in[21]; const float* ec_b1= (const float*)d_in[22];
  const float* ec_w2= (const float*)d_in[23]; const float* ec_b2= (const float*)d_in[24];
  const float* nc_w1= (const float*)d_in[25]; const float* nc_b1= (const float*)d_in[26];
  const float* nc_w2= (const float*)d_in[27]; const float* nc_b2= (const float*)d_in[28];
  const float* cl_w1= (const float*)d_in[29]; const float* cl_b1= (const float*)d_in[30];
  const float* cl_w2= (const float*)d_in[31]; const float* cl_b2= (const float*)d_in[32];

  const int* src = ei;        // edge_index[0]
  const int* tgt = ei + EE;   // edge_index[1]

  char* ws = (char*)d_ws;
  size_t off = 0;
  auto alloc = [&](size_t bytes) { char* p = ws + off; off += (bytes + 255) & ~(size_t)255; return p; };

  half_t* x16    = (half_t*)alloc((size_t)NN * 256 * 2);
  half_t* ea16   = (half_t*)alloc((size_t)EE * 64 * 2);
  half_t* wne16  = (half_t*)alloc((size_t)128 * 256 * 2);
  half_t* wee16  = (half_t*)alloc((size_t)128 * 64 * 2);
  half_t* wme116 = (half_t*)alloc((size_t)256 * 416 * 2);
  half_t* wme216 = (half_t*)alloc((size_t)128 * 256 * 2);
  half_t* wmn16  = (half_t*)alloc((size_t)128 * 288 * 2);   // K padded 272 -> 288
  float*  qbuf   = (float*)alloc((size_t)BB * 16 * PP * 4);
  float*  vbuf   = (float*)alloc((size_t)BB * 16 * PP * 4);
  float*  nfA    = (float*)alloc((size_t)NN * DD * 4);
  float*  nfB    = (float*)alloc((size_t)NN * DD * 4);
  half_t* nf16   = (half_t*)alloc((size_t)NN * 144 * 2);    // [nf | res]
  float*  ef32   = (float*)alloc((size_t)EE * EDD * 4);
  half_t* ef16   = (half_t*)alloc((size_t)EE * EDD * 2);
  half_t* h16    = (half_t*)alloc((size_t)EE * HIDD * 2);

  auto cvt = [&](const float* in, half_t* out, int n) {
    k_f32_to_f16<<<(n + 255) / 256, 256, 0, stream>>>(in, out, n);
  };
  cvt(x, x16, NN * 256);
  cvt(ea, ea16, EE * 64);
  cvt(ne_w, wne16, 128 * 256);
  cvt(ee_w, wee16, 128 * 64);
  cvt(me_w1, wme116, 256 * 416);
  cvt(me_w2, wme216, 128 * 256);
  k_f32_to_f16_pad<<<(128 * 288 + 255) / 256, 256, 0, stream>>>(mn_w, wmn16, 128, 272, 288);

  k_qv<<<(BB * 16 * PP) / 256, 256, 0, stream>>>(fm, q_w, q_b, v_w, v_b, qbuf, vbuf);

  // node encoder: nfA = relu(x @ ne_w^T + ne_b)           M=2048 K=256 N=128
  k_gemm<256, 256, 256, false, false, false, false, 0><<<dim3(NN / 32, 2), 32, 0, stream>>>(
      x16, 256, nullptr, nullptr, 0, nullptr, nullptr, 0,
      wne16, ne_b, 128, nfA, nullptr, nullptr);
  // edge encoder: ef = relu(edge_attr @ ee_w^T + ee_b)    M=65536 K=64 N=128
  k_gemm<64, 64, 64, false, false, false, false, 1><<<dim3(EE / 32, 2), 32, 0, stream>>>(
      ea16, 64, nullptr, nullptr, 0, nullptr, nullptr, 0,
      wee16, ee_b, 128, ef32, ef16, nullptr);

  float* nf_cur = nfA; float* nf_nxt = nfB;
  for (int step = 0; step < 4; ++step) {
    k_attention<<<NN, 256, 0, stream>>>(nf_cur, key_w, key_b, qbuf, vbuf, bix, nf16);
    k_pack_nf<<<(NN * DD) / 256, 256, 0, stream>>>(nf_cur, nf16);
    // edge MLP layer 1: A = [nf16[tgt] | nf16[src] | ef16]   M=65536 K=416 N=256
    k_gemm<416, 144, 288, true, true, true, true, 2><<<dim3(EE / 32, 4), 32, 0, stream>>>(
        nf16, 144, tgt, nf16, 144, src, ef16, 128,
        wme116, me_b1, 256, nullptr, h16, nullptr);
    // edge MLP layer 2: ef = relu(h @ me_w2^T + me_b2)       M=65536 K=256 N=128
    k_gemm<256, 256, 256, false, false, false, false, 1><<<dim3(EE / 32, 2), 32, 0, stream>>>(
        h16, 256, nullptr, nullptr, 0, nullptr, nullptr, 0,
        wme216, me_b2, 128, ef32, ef16, nullptr);
    // message + fused segment_sum: A = [nf16[tgt] | ef16 | 0-pad]  M=65536 K=288 N=128
    hipMemsetAsync(nf_nxt, 0, (size_t)NN * DD * 4, stream);
    k_gemm<288, 144, 272, true, false, true, false, 3><<<dim3(EE / 32, 2), 32, 0, stream>>>(
        nf16, 144, tgt, ef16, 128, nullptr, nullptr, 0,
        wmn16, mn_b, 128, nf_nxt, nullptr, tgt);
    float* tmp = nf_cur; nf_cur = nf_nxt; nf_nxt = tmp;
  }

  // output layout: pe[E], pn[N], pc[N,17], node_features[N,128], edge_features[E,128]
  float* pe = (float*)d_out;
  float* pn = pe + EE;
  float* pc = pn + NN;
  float* nf_out = pc + (size_t)NN * 17;
  float* ef_out = nf_out + (size_t)NN * DD;
  hipMemcpyAsync(nf_out, nf_cur, (size_t)NN * DD * 4, hipMemcpyDeviceToDevice, stream);
  hipMemcpyAsync(ef_out, ef32, (size_t)EE * EDD * 4, hipMemcpyDeviceToDevice, stream);
  k_edge_readout<<<EE / 128, 128, 0, stream>>>(ef32, ec_w1, ec_b1, ec_w2, ec_b2, pe);
  k_node_readout<<<NN / 128, 128, 0, stream>>>(nf_cur, nc_w1, nc_b1, nc_w2, nc_b2,
                                               cl_w1, cl_b1, cl_w2, cl_b2, pn, pc);
}
